// CauseMutiHeadAttention_14774687498245
// MI455X (gfx1250) — matrix operand order repaired and checked by reading, not compiled, not run on hardware
//
#include <hip/hip_runtime.h>

typedef __attribute__((ext_vector_type(16))) __bf16 v16bf;
typedef __attribute__((ext_vector_type(8)))  float  v8f;
typedef __attribute__((ext_vector_type(4)))  int    v4i;

#define S_LEN  2048
#define D_MOD  1024
#define M_TOK  4096

#if __has_builtin(__builtin_amdgcn_global_load_async_to_lds_b128) && \
    __has_builtin(__builtin_amdgcn_s_wait_asynccnt)
#define ASYNC_LDS 1
#endif

__device__ __forceinline__ unsigned short f2bf(float f) {
  unsigned u = __builtin_bit_cast(unsigned, f);
  u += 0x7fffu + ((u >> 16) & 1u);          // round-to-nearest-even
  return (unsigned short)(u >> 16);
}

__device__ __forceinline__ v8f zero8() {
  v8f z;
#pragma unroll
  for (int i = 0; i < 8; ++i) z[i] = 0.0f;
  return z;
}

union FragAB {                 // 16 bf16 = 8 VGPRs
  uint4 q[2];
  v16bf v;
};

__device__ __forceinline__ v8f wmma_bf16(const FragAB& a, const FragAB& b, v8f c) {
  return __builtin_amdgcn_wmma_f32_16x16x32_bf16(
      false, a.v, false, b.v, (short)0, c, false, false);
}

// 16-byte global -> LDS copy; async (ASYNCcnt) when the toolchain exposes it.
__device__ __forceinline__ void cp16(void* lds, const void* g) {
#ifdef ASYNC_LDS
  __builtin_amdgcn_global_load_async_to_lds_b128(
      (__attribute__((address_space(1))) v4i*)g,
      (__attribute__((address_space(3))) v4i*)lds, 0, 0);
#else
  *(uint4*)lds = *(const uint4*)g;
#endif
}

__device__ __forceinline__ void cp16_wait() {
#ifdef ASYNC_LDS
  __builtin_amdgcn_s_wait_asynccnt(0);
#endif
}

// ---------------------------------------------------------------- fp32 -> bf16
__global__ void cvt_bf16_kernel(const float* __restrict__ src,
                                unsigned short* __restrict__ dst, int n) {
  int i = blockIdx.x * blockDim.x + threadIdx.x;
  if (i < n) dst[i] = f2bf(src[i]);
}

// ---------------------------------------------------------------- GEMM
// Y[m,e] = sum_k A[m,k] * W[e,k] + bias[e]     (A,W bf16; accum fp32)
// mode 0: fp32 store to outF[m*1024+e]
// mode 1: RoPE (angle from tpos[m]) then bf16 store [b,h,s,64], *scale
// mode 2: bf16 transposed store [b,h,64,s]
__global__ __launch_bounds__(256) void gemm_bf16_kernel(
    const unsigned short* __restrict__ A,
    const unsigned short* __restrict__ W,
    const float* __restrict__ bias,
    float* __restrict__ outF,
    unsigned short* __restrict__ outB,
    const int* __restrict__ tpos,
    int mode, float scale) {
  __shared__ uint4 ldsA[2][512];       // 2 x (128 x 32 bf16)
  __shared__ uint4 ldsB[2][512];

  const int tid  = threadIdx.x;
  const int lane = tid & 31;
  const int half = lane >> 4;
  const int wv   = tid >> 5;
  const int wm   = wv >> 2;         // 0..1 -> 64-row slab
  const int wn   = wv & 3;          // 0..3 -> 32-col slab
  const int mBase = blockIdx.y * 128;
  const int nBase = blockIdx.x * 128;

  const uint4* A4 = (const uint4*)A;
  const uint4* W4 = (const uint4*)W;

  auto loadTile = [&](int k, int buf) {
#pragma unroll
    for (int i = 0; i < 2; ++i) {
      int idx = tid + i * 256;
      int row = idx >> 2, c = idx & 3;
      cp16(&ldsA[buf][idx], &A4[(size_t)(mBase + row) * 128 + k * 4 + c]);
      cp16(&ldsB[buf][idx], &W4[(size_t)(nBase + row) * 128 + k * 4 + c]);
    }
  };

  v8f acc[4][2];
#pragma unroll
  for (int mt = 0; mt < 4; ++mt)
#pragma unroll
    for (int nt = 0; nt < 2; ++nt) acc[mt][nt] = zero8();

  loadTile(0, 0);
  cp16_wait();
  __syncthreads();

  for (int k = 0; k < D_MOD / 32; ++k) {
    const int buf = k & 1;
    if (k + 1 < D_MOD / 32) loadTile(k + 1, buf ^ 1);

    FragAB fa[4], fb[2];
#pragma unroll
    for (int mt = 0; mt < 4; ++mt) {
      int row = wm * 64 + mt * 16 + (lane & 15);
      fa[mt].q[0] = ldsA[buf][row * 4 + half];
      fa[mt].q[1] = ldsA[buf][row * 4 + 2 + half];
    }
#pragma unroll
    for (int nt = 0; nt < 2; ++nt) {
      int row = wn * 32 + nt * 16 + (lane & 15);
      fb[nt].q[0] = ldsB[buf][row * 4 + half];
      fb[nt].q[1] = ldsB[buf][row * 4 + 2 + half];
    }
#pragma unroll
    for (int mt = 0; mt < 4; ++mt)
#pragma unroll
      for (int nt = 0; nt < 2; ++nt)
        acc[mt][nt] = wmma_bf16(fa[mt], fb[nt], acc[mt][nt]);

    cp16_wait();          // our async writes into buf^1 are done
    __syncthreads();      // everyone finished reading buf + writing buf^1
  }

  // ------------------------------------------------------------- epilogue
#pragma unroll
  for (int mt = 0; mt < 4; ++mt) {
#pragma unroll
    for (int nt = 0; nt < 2; ++nt) {
#pragma unroll
      for (int v = 0; v < 8; ++v) {
        int m = mBase + wm * 64 + mt * 16 + v + half * 8;
        int e = nBase + wn * 32 + nt * 16 + (lane & 15);
        float val = acc[mt][nt][v] + bias[e];
        if (mode == 0) {
          outF[(size_t)m * D_MOD + e] = val;
        } else if (mode == 1) {
          int pos  = tpos[m];
          int slot = m & (S_LEN - 1);
          int b    = m >> 11;
          int head = e >> 6, idx = e & 63;
          int p    = idx >> 1;
          // inv_freq = 10000^(-2p/64) = exp(-2p/64 * ln 1e4)
          float inv = __expf(-(float)(2 * p) * (9.210340371976184f / 64.0f));
          float sn, cs;
          __sincosf((float)pos * inv, &sn, &cs);
          float other = __shfl_xor(val, 1);
          float out = val * cs + other * sn * (((idx & 1) == 0) ? -1.0f : 1.0f);
          outB[((size_t)(b * 16 + head) * S_LEN + slot) * 64 + idx] = f2bf(out * scale);
        } else {
          int slot = m & (S_LEN - 1);
          int b    = m >> 11;
          int head = e >> 6, idx = e & 63;
          outB[((size_t)(b * 16 + head) * 64 + idx) * S_LEN + slot] = f2bf(val);
        }
      }
    }
  }
}

// ---------------------------------------------------------------- flash attention
// Q,K: [b,h,s,64] bf16 (Q pre-scaled 1/8);  V: [b,h,64,s] bf16 (transposed)
// O:   [b*s, 1024] bf16 (heads re-interleaved for the final projection)
__global__ __launch_bounds__(256) void attn_kernel(
    const unsigned short* __restrict__ Q,
    const unsigned short* __restrict__ K,
    const unsigned short* __restrict__ V,
    unsigned short* __restrict__ O) {
  __shared__ uint4 ldsK[1024];                        // 128 x 64 bf16  (16KB)
  __shared__ uint4 ldsV[1024];                        // 64 x 128 bf16  (16KB)
  __shared__ __align__(16) unsigned short ldsP[8 * 16 * 128];  // 32KB

  const int tid  = threadIdx.x;
  const int lane = tid & 31;
  const int half = lane >> 4;
  const int wv   = tid >> 5;
  const int qb   = blockIdx.x;
  const int bh   = blockIdx.y;
  const int qBase = qb * 128;

  const uint4* Q4 = (const uint4*)Q;
  const uint4* K4 = (const uint4*)K;
  const uint4* V4 = (const uint4*)V;
  const uint4* P4 = (const uint4*)ldsP;
  unsigned short* Pw = ldsP + wv * (16 * 128);

  // Q fragments for this wave's 16 rows, 2 K-steps over d_k=64
  FragAB qf[2];
  {
    int row = qBase + wv * 16 + (lane & 15);
    size_t rb = ((size_t)bh * S_LEN + row) * 8;       // 8 uint4 per 64-bf16 row
#pragma unroll
    for (int ks = 0; ks < 2; ++ks) {
      qf[ks].q[0] = Q4[rb + ks * 4 + half];
      qf[ks].q[1] = Q4[rb + ks * 4 + 2 + half];
    }
  }

  float mrow[8], lsum[8];
#pragma unroll
  for (int v = 0; v < 8; ++v) { mrow[v] = -1e30f; lsum[v] = 0.0f; }
  v8f ao[4];
#pragma unroll
  for (int dt = 0; dt < 4; ++dt) ao[dt] = zero8();

  for (int kt = 0; kt <= qb; ++kt) {
    const int kBase = kt * 128;
    __syncthreads();
#pragma unroll
    for (int i = 0; i < 4; ++i) {
      int idx = tid + i * 256;
      { int row = idx >> 3, c = idx & 7;
        cp16(&ldsK[idx], &K4[((size_t)bh * S_LEN + kBase + row) * 8 + c]); }
      { int row = idx >> 4, c = idx & 15;
        cp16(&ldsV[idx], &V4[((size_t)bh * 64 + row) * 256 + (kBase >> 3) + c]); }
    }
    cp16_wait();
    __syncthreads();

    // scores: 16 rows x 128 keys
    v8f st[8];
#pragma unroll
    for (int nt = 0; nt < 8; ++nt) st[nt] = zero8();
#pragma unroll
    for (int nt = 0; nt < 8; ++nt) {
      int krow = nt * 16 + (lane & 15);
#pragma unroll
      for (int ks = 0; ks < 2; ++ks) {
        FragAB kb;
        kb.q[0] = ldsK[krow * 8 + ks * 4 + half];
        kb.q[1] = ldsK[krow * 8 + ks * 4 + 2 + half];
        st[nt] = wmma_bf16(qf[ks], kb, st[nt]);
      }
    }

    if (kt == qb) {               // causal mask only on the diagonal tile
#pragma unroll
      for (int nt = 0; nt < 8; ++nt)
#pragma unroll
        for (int v = 0; v < 8; ++v) {
          int key = kBase + nt * 16 + (lane & 15);
          int qr  = qBase + wv * 16 + v + half * 8;
          if (key > qr) st[nt][v] = -1e30f;
        }
    }

    // online softmax (fp32), per row v of this wave
#pragma unroll
    for (int v = 0; v < 8; ++v) {
      float mx = st[0][v];
#pragma unroll
      for (int nt = 1; nt < 8; ++nt) mx = fmaxf(mx, st[nt][v]);
      mx = fmaxf(mx, __shfl_xor(mx, 1));
      mx = fmaxf(mx, __shfl_xor(mx, 2));
      mx = fmaxf(mx, __shfl_xor(mx, 4));
      mx = fmaxf(mx, __shfl_xor(mx, 8));
      float mnew  = fmaxf(mrow[v], mx);
      float alpha = __expf(mrow[v] - mnew);
      float rsum = 0.0f;
#pragma unroll
      for (int nt = 0; nt < 8; ++nt) {
        float p = __expf(st[nt][v] - mnew);
        st[nt][v] = p;
        rsum += p;
      }
      rsum += __shfl_xor(rsum, 1);
      rsum += __shfl_xor(rsum, 2);
      rsum += __shfl_xor(rsum, 4);
      rsum += __shfl_xor(rsum, 8);
      mrow[v] = mnew;
      lsum[v] = lsum[v] * alpha + rsum;
#pragma unroll
      for (int dt = 0; dt < 4; ++dt) ao[dt][v] *= alpha;
    }

    // bounce P through per-wave LDS: C-layout -> A-layout (bf16)
#pragma unroll
    for (int nt = 0; nt < 8; ++nt)
#pragma unroll
      for (int v = 0; v < 8; ++v)
        Pw[(v + half * 8) * 128 + nt * 16 + (lane & 15)] = f2bf(st[nt][v]);

    // PV: 4 K-steps of 32 keys  x  4 d_k tiles
#pragma unroll
    for (int ks = 0; ks < 4; ++ks) {
      FragAB pf;
      int prow4 = wv * 256 + (lane & 15) * 16;   // uint4 units
      pf.q[0] = P4[prow4 + ks * 4 + half];
      pf.q[1] = P4[prow4 + ks * 4 + 2 + half];
#pragma unroll
      for (int dt = 0; dt < 4; ++dt) {
        FragAB vf;
        int vrow = dt * 16 + (lane & 15);
        vf.q[0] = ldsV[vrow * 16 + ks * 4 + half];
        vf.q[1] = ldsV[vrow * 16 + ks * 4 + 2 + half];
        ao[dt] = wmma_bf16(pf, vf, ao[dt]);
      }
    }
  }

  // write attn output in [b*s, h*64+dk] layout for the O-projection
  int b = bh >> 4, head = bh & 15;
#pragma unroll
  for (int dt = 0; dt < 4; ++dt)
#pragma unroll
    for (int v = 0; v < 8; ++v) {
      int dk  = dt * 16 + (lane & 15);
      int row = qBase + wv * 16 + v + half * 8;
      float val = ao[dt][v] / lsum[v];
      O[((size_t)(b * S_LEN + row)) * D_MOD + head * 64 + dk] = f2bf(val);
    }
}

// ---------------------------------------------------------------- launch
extern "C" void kernel_launch(void* const* d_in, const int* in_sizes, int n_in,
                              void* d_out, int out_size, void* d_ws, size_t ws_size,
                              hipStream_t stream) {
  const float* x  = (const float*)d_in[0];
  const float* Wq = (const float*)d_in[1];
  const float* bq = (const float*)d_in[2];
  const float* Wk = (const float*)d_in[3];
  const float* bk = (const float*)d_in[4];
  const float* Wv = (const float*)d_in[5];
  const float* bv = (const float*)d_in[6];
  const float* Wo = (const float*)d_in[7];
  const float* bo = (const float*)d_in[8];
  const int*   tp = (const int*)d_in[9];

  char* ws = (char*)d_ws;
  const size_t MB = (size_t)1 << 20;
  unsigned short* xb  = (unsigned short*)(ws + 0 * MB);   // x bf16          8MB
  unsigned short* wqb = (unsigned short*)(ws + 8 * MB);   // Wq bf16         2MB
  unsigned short* wkb = (unsigned short*)(ws + 10 * MB);
  unsigned short* wvb = (unsigned short*)(ws + 12 * MB);
  unsigned short* wob = (unsigned short*)(ws + 14 * MB);
  unsigned short* qbf = (unsigned short*)(ws + 16 * MB);  // Q [b,h,s,64]    8MB
  unsigned short* kbf = (unsigned short*)(ws + 24 * MB);  // K [b,h,s,64]    8MB
  unsigned short* vtb = (unsigned short*)(ws + 32 * MB);  // V [b,h,64,s]    8MB
  unsigned short* atb = (unsigned short*)(ws + 40 * MB);  // attn bf16       8MB

  cvt_bf16_kernel<<<(M_TOK * D_MOD) / 256, 256, 0, stream>>>(x, xb, M_TOK * D_MOD);
  cvt_bf16_kernel<<<(D_MOD * D_MOD) / 256, 256, 0, stream>>>(Wq, wqb, D_MOD * D_MOD);
  cvt_bf16_kernel<<<(D_MOD * D_MOD) / 256, 256, 0, stream>>>(Wk, wkb, D_MOD * D_MOD);
  cvt_bf16_kernel<<<(D_MOD * D_MOD) / 256, 256, 0, stream>>>(Wv, wvb, D_MOD * D_MOD);
  cvt_bf16_kernel<<<(D_MOD * D_MOD) / 256, 256, 0, stream>>>(Wo, wob, D_MOD * D_MOD);

  dim3 gg(8, 32), gb(256);
  // Q: RoPE + 1/sqrt(64) folded in (exact power-of-two scale in bf16)
  gemm_bf16_kernel<<<gg, gb, 0, stream>>>(xb, wqb, bq, nullptr, qbf, tp, 1, 0.125f);
  gemm_bf16_kernel<<<gg, gb, 0, stream>>>(xb, wkb, bk, nullptr, kbf, tp, 1, 1.0f);
  gemm_bf16_kernel<<<gg, gb, 0, stream>>>(xb, wvb, bv, nullptr, vtb, tp, 2, 1.0f);

  attn_kernel<<<dim3(16, 32), 256, 0, stream>>>(qbf, kbf, vtb, atb);

  gemm_bf16_kernel<<<gg, gb, 0, stream>>>(atb, wob, bo, (float*)d_out, nullptr, tp, 0, 1.0f);
}
